// LigerFusedLinearCrossEntropyHelion_39230231282355
// MI455X (gfx1250) — matrix operand order repaired and checked by reading, not compiled, not run on hardware
//
#include <hip/hip_runtime.h>
#include <hip/hip_bf16.h>

// ---------------------------------------------------------------------------
// Fused linear cross-entropy for MI455X (gfx1250, wave32, WMMA).
//   logits = input[4096,2048] @ weight[32000,2048]^T  (bf16 WMMA, f32 accum)
//   out    = sum_i (lse_i - tgt_logit_i) / n_valid    (online logsumexp fused)
// Weight is pre-converted once to bf16 in WMMA-B fragment order inside d_ws,
// so the hot loop streams 8.4 GB instead of 16 GB and does zero cvt VALU.
// ---------------------------------------------------------------------------

#define BT 4096
#define H  2048
#define V  32000
#define IGNORE_INDEX (-100)

#define ROWS_PER_WG 64          // M tile held in LDS as bf16
#define NSPLITS 8               // V split across grid.y
#define VSPLIT (V / NSPLITS)    // 4000
#define NCHUNKS (VSPLIT / 16)   // 250 16-wide vocab chunks per split
#define KSTEPS (H / 32)         // 64 WMMA K-steps of 32
#define APITCH 2056             // bf16 elements per row in LDS (pad vs banks)

#define TOTAL_CHUNKS (V / 16)            // 2000
#define WBF_ELEMS ((size_t)V * H)        // 65,536,000 bf16
#define WBF_BYTES (WBF_ELEMS * 2)        // 131,072,000 bytes
#define PART_FLOATS (3 * NSPLITS * BT + 2)

typedef __attribute__((ext_vector_type(16))) __bf16 v16bf;
typedef __attribute__((ext_vector_type(8)))  __bf16 v8bf;
typedef __attribute__((ext_vector_type(4)))  __bf16 v4bf;
typedef __attribute__((ext_vector_type(8)))  float  v8f;

__device__ __forceinline__ __bf16 to_bf16(float f) {
    unsigned u = __builtin_bit_cast(unsigned, f);
    unsigned r = (u + 0x7FFFu + ((u >> 16) & 1u)) >> 16;
    return __builtin_bit_cast(__bf16, (unsigned short)r);
}

// Partial-buffer float layout (base pointer `part`):
//   [0,     32768)  pm : per (split,row) running max
//   [32768, 65536)  ps : per (split,row) sum of exp
//   [65536, 98304)  pt : per (split,row) target logit (0 if not in split)
//   [98304] nll sum   [98305] valid count
#define P_PM 0
#define P_PS (NSPLITS * BT)
#define P_PT (2 * NSPLITS * BT)
#define P_ACC (3 * NSPLITS * BT)

__global__ void flce_init_kernel(float* part) {
    part[P_ACC + 0] = 0.0f;
    part[P_ACC + 1] = 0.0f;
}

// One-shot: fp32 weight -> bf16, stored in WMMA-B fragment order.
// frag f = (chunk*KSTEPS + kc)*32 + lane holds 16 bf16:
//   row = chunk*16 + (lane&15), cols kc*32 + (lane>>4)*16 + [0..16)
__global__ __launch_bounds__(256)
void flce_wconv_kernel(const float* __restrict__ weight,
                       __bf16* __restrict__ wbf) {
    size_t idx8 = (size_t)blockIdx.x * 256 + threadIdx.x;  // one 8-elem half-frag
    int sub  = (int)(idx8 & 1);
    int lane = (int)((idx8 >> 1) & 31);
    int kc   = (int)((idx8 >> 6) & 63);
    int c    = (int)(idx8 >> 12);
    int row  = c * 16 + (lane & 15);
    int col  = kc * 32 + (lane >> 4) * 16 + sub * 8;
    const float4* src = (const float4*)(weight + (size_t)row * H + col);
    float4 f0 = src[0], f1 = src[1];
    v8bf b;
    b[0] = to_bf16(f0.x); b[1] = to_bf16(f0.y);
    b[2] = to_bf16(f0.z); b[3] = to_bf16(f0.w);
    b[4] = to_bf16(f1.x); b[5] = to_bf16(f1.y);
    b[6] = to_bf16(f1.z); b[7] = to_bf16(f1.w);
    *(v8bf*)(wbf + idx8 * 8) = b;
}

template <bool PRECONV>
__global__ __launch_bounds__(256)
void flce_partial_kernel(const float*  __restrict__ input,
                         const float*  __restrict__ weight_f32,
                         const __bf16* __restrict__ wbf,
                         const int*    __restrict__ target,
                         float* __restrict__ part) {
    extern __shared__ char smem[];
    __bf16* Abuf = (__bf16*)smem;                                   // 64*APITCH bf16
    int*    tgtL = (int*)  (smem + ROWS_PER_WG * APITCH * 2);       // 64 ints
    float*  lpm  = (float*)(smem + ROWS_PER_WG * APITCH * 2 + 256); // 64*8 x3
    float*  lps  = lpm + ROWS_PER_WG * 8;
    float*  lpt  = lps + ROWS_PER_WG * 8;

    const int tid     = threadIdx.x;
    const int rowBase = blockIdx.x * ROWS_PER_WG;
    const int split   = blockIdx.y;

    // ---- Phase A: stage 64x2048 input rows into LDS as bf16 ----
    {
        const float4* in4 = (const float4*)(input + (size_t)rowBase * H);
        for (int i = tid; i < ROWS_PER_WG * (H / 4); i += 256) {
            int row = i >> 9;            // H/4 = 512 float4 per row
            int k4  = i & 511;
            float4 f = in4[(size_t)row * (H / 4) + k4];
            v4bf b;
            b[0] = to_bf16(f.x); b[1] = to_bf16(f.y);
            b[2] = to_bf16(f.z); b[3] = to_bf16(f.w);
            *(v4bf*)(Abuf + (size_t)row * APITCH + k4 * 4) = b;
        }
        if (tid < ROWS_PER_WG) tgtL[tid] = target[rowBase + tid];
    }
    __syncthreads();

    const int w    = tid >> 5;     // wave id (8 waves)
    const int lane = tid & 31;
    const int half = lane >> 4;
    const int ln   = lane & 15;

    // Online softmax state: 4 M-tiles x 8 C-rows per lane.
    float mRun[4][8], sRun[4][8], tAcc[4][8];
#pragma unroll
    for (int mt = 0; mt < 4; ++mt)
#pragma unroll
        for (int r = 0; r < 8; ++r) {
            mRun[mt][r] = -1e30f; sRun[mt][r] = 0.0f; tAcc[mt][r] = 0.0f;
        }

    // ---- Phase B: stream bf16 weight fragments, WMMA, fold logsumexp ----
    for (int chunk = w; chunk < NCHUNKS; chunk += 8) {
        const int cGlobal = split * NCHUNKS + chunk;
        const int vcol0   = cGlobal * 16;

        const __bf16* wfrag = nullptr;
        const float*  wbase = nullptr;
        if (PRECONV) {
            wfrag = wbf + ((size_t)cGlobal * KSTEPS * 32 + lane) * 16;
        } else {
            wbase = weight_f32 + (size_t)(vcol0 + ln) * H + half * 16;
        }

        v8f acc[4];
#pragma unroll
        for (int mt = 0; mt < 4; ++mt) acc[mt] = (v8f)0.0f;

        for (int kc = 0; kc < KSTEPS; ++kc) {
            const int k0 = kc * 32;
            v16bf b;
            if (PRECONV) {
                b = *(const v16bf*)(wfrag + (size_t)kc * 512);  // 32B coalesced
            } else {
                const float4* wp = (const float4*)(wbase + k0);
                float4 f0 = wp[0], f1 = wp[1], f2 = wp[2], f3 = wp[3];
                b[0]  = to_bf16(f0.x); b[1]  = to_bf16(f0.y);
                b[2]  = to_bf16(f0.z); b[3]  = to_bf16(f0.w);
                b[4]  = to_bf16(f1.x); b[5]  = to_bf16(f1.y);
                b[6]  = to_bf16(f1.z); b[7]  = to_bf16(f1.w);
                b[8]  = to_bf16(f2.x); b[9]  = to_bf16(f2.y);
                b[10] = to_bf16(f2.z); b[11] = to_bf16(f2.w);
                b[12] = to_bf16(f3.x); b[13] = to_bf16(f3.y);
                b[14] = to_bf16(f3.z); b[15] = to_bf16(f3.w);
            }

#pragma unroll
            for (int mt = 0; mt < 4; ++mt) {
                // A layout: lane row = ln; elems 0..15 K=k0+16*half..
                const __bf16* ap = Abuf + (size_t)(mt * 16 + ln) * APITCH + k0 + half * 16;
                v8bf lo = *(const v8bf*)ap;
                v8bf hi = *(const v8bf*)(ap + 8);
                v16bf a = __builtin_shufflevector(lo, hi,
                        0, 1, 2, 3, 4, 5, 6, 7, 8, 9, 10, 11, 12, 13, 14, 15);
                acc[mt] = __builtin_amdgcn_wmma_f32_16x16x32_bf16(
                        false, a, false, b, (short)0, acc[mt], false, false);
            }
        }

        // Fold the 16x16 tiles into the running (max, sumexp, tgt) state.
        const int col = vcol0 + ln;
#pragma unroll
        for (int mt = 0; mt < 4; ++mt)
#pragma unroll
            for (int r = 0; r < 8; ++r) {
                float x = acc[mt][r];
                int rowl = mt * 16 + half * 8 + r;
                if (tgtL[rowl] == col) tAcc[mt][r] += x;
                float nm = fmaxf(mRun[mt][r], x);
                sRun[mt][r] = sRun[mt][r] * __expf(mRun[mt][r] - nm) + __expf(x - nm);
                mRun[mt][r] = nm;
            }
    }

    // ---- Cross-lane merge over the 16 columns of each half-wave group ----
#pragma unroll
    for (int mt = 0; mt < 4; ++mt)
#pragma unroll
        for (int r = 0; r < 8; ++r) {
            float m = mRun[mt][r], s = sRun[mt][r], t = tAcc[mt][r];
            for (int off = 8; off; off >>= 1) {
                float om = __shfl_xor(m, off, 16);
                float os = __shfl_xor(s, off, 16);
                t += __shfl_xor(t, off, 16);
                float nm = fmaxf(m, om);
                s = s * __expf(m - nm) + os * __expf(om - nm);
                m = nm;
            }
            if (ln == 0) {
                int rowl = mt * 16 + half * 8 + r;
                lpm[rowl * 8 + w] = m;
                lps[rowl * 8 + w] = s;
                lpt[rowl * 8 + w] = t;
            }
        }
    __syncthreads();

    // ---- Cross-wave merge, write per-(split,row) partial to workspace ----
    if (tid < ROWS_PER_WG) {
        float m = -1e30f, s = 0.0f, t = 0.0f;
#pragma unroll
        for (int w2 = 0; w2 < 8; ++w2) {
            float om = lpm[tid * 8 + w2];
            float os = lps[tid * 8 + w2];
            t += lpt[tid * 8 + w2];
            float nm = fmaxf(m, om);
            s = s * __expf(m - nm) + os * __expf(om - nm);
            m = nm;
        }
        size_t g = (size_t)split * BT + rowBase + tid;
        part[P_PM + g] = m;
        part[P_PS + g] = s;
        part[P_PT + g] = t;
    }
}

__global__ __launch_bounds__(256)
void flce_reduce_kernel(const int* __restrict__ target,
                        float* __restrict__ part) {
    int row = blockIdx.x * blockDim.x + threadIdx.x;
    if (row >= BT) return;
    float m = -1e30f, s = 0.0f, t = 0.0f;
#pragma unroll
    for (int sp = 0; sp < NSPLITS; ++sp) {
        size_t g = (size_t)sp * BT + row;
        float om = part[P_PM + g];
        float os = part[P_PS + g];
        t += part[P_PT + g];
        float nm = fmaxf(m, om);
        s = s * __expf(m - nm) + os * __expf(om - nm);
        m = nm;
    }
    float lse = m + __logf(s);
    float nll = lse - t;                       // t == 0 when target ignored
    float valid = (target[row] != IGNORE_INDEX) ? 1.0f : 0.0f;
    atomicAdd(&part[P_ACC + 0], nll);
    atomicAdd(&part[P_ACC + 1], valid);
}

__global__ void flce_final_kernel(const float* __restrict__ part,
                                  float* __restrict__ out) {
    out[0] = part[P_ACC + 0] / part[P_ACC + 1];
}

extern "C" void kernel_launch(void* const* d_in, const int* in_sizes, int n_in,
                              void* d_out, int out_size, void* d_ws, size_t ws_size,
                              hipStream_t stream) {
    const float* input  = (const float*)d_in[0];
    const float* weight = (const float*)d_in[1];
    const int*   target = (const int*)d_in[2];
    float* out = (float*)d_out;

    const bool preconv = ws_size >= WBF_BYTES + (size_t)PART_FLOATS * 4;
    __bf16* wbf = (__bf16*)d_ws;
    float*  part = preconv ? (float*)((char*)d_ws + WBF_BYTES) : (float*)d_ws;

    // 64x2048 bf16 A tile + targets + cross-wave partials (CDNA5 WGP: 320KB LDS)
    size_t smem = (size_t)ROWS_PER_WG * APITCH * 2 + 256 + 3 * ROWS_PER_WG * 8 * 4;
    static int attr_set = 0;
    if (!attr_set) {
        hipFuncSetAttribute((const void*)flce_partial_kernel<true>,
                            hipFuncAttributeMaxDynamicSharedMemorySize, (int)smem);
        hipFuncSetAttribute((const void*)flce_partial_kernel<false>,
                            hipFuncAttributeMaxDynamicSharedMemorySize, (int)smem);
        attr_set = 1;
    }

    flce_init_kernel<<<1, 1, 0, stream>>>(part);

    dim3 grid(BT / ROWS_PER_WG, NSPLITS);     // 64 x 8
    if (preconv) {
        // 8,192,000 half-fragments of 8 bf16 -> 32000 blocks of 256 threads
        flce_wconv_kernel<<<(int)(WBF_ELEMS / 8 / 256), 256, 0, stream>>>(weight, wbf);
        flce_partial_kernel<true><<<grid, 256, smem, stream>>>(input, weight, wbf,
                                                               target, part);
    } else {
        flce_partial_kernel<false><<<grid, 256, smem, stream>>>(input, weight, wbf,
                                                                target, part);
    }

    flce_reduce_kernel<<<BT / 256, 256, 0, stream>>>(target, part);

    flce_final_kernel<<<1, 1, 0, stream>>>(out ? part : part, out);
}
